// SparseBertSelfAttention_13675175870905
// MI455X (gfx1250) — matrix operand order repaired and checked by reading, not compiled, not run on hardware
//
#include <hip/hip_runtime.h>
#include <hip/hip_bf16.h>
#include <math.h>

// Problem sizes (fixed by reference)
#define SEQ 1024
#define BATCH 4
#define HIDDEN 1024
#define HEADS 16
#define HEAD_DIM 64
#define ROWS (SEQ * BATCH)   // 4096 flattened (s,b) rows

typedef __attribute__((ext_vector_type(16))) _Float16 v16h;
typedef __attribute__((ext_vector_type(8)))  _Float16 v8h;
typedef __attribute__((ext_vector_type(8)))  float    v8f;

// ---------------------------------------------------------------------------
// Fragment loaders (WMMA 16x16x32 f16, wave32; contiguous k order used alike for A and B).
//
// A (16x32, MxK): lane holds row m = lane%16, g = lane/16.
//   comps[0..15] <- K = k0 + 16g + (0..15)    = p[0..15]
//   with p = &src[row*ld + k0 + 16g]  -> two 16B loads.
// B (32x16, KxN): lane holds col n = lane%16.
//   comps[0..15] <- K = k0 + 16g + (0..15)    = p[0..15]
//   with p = &src[n*ld + k0 + 16g]    -> two 16B loads.
// ---------------------------------------------------------------------------

__device__ __forceinline__ v16h fragA(const _Float16* p) {
  v16h a;
#pragma unroll
  for (int i = 0; i < 16; ++i) a[i] = p[i];
  return a;
}

__device__ __forceinline__ v16h fragB(const _Float16* p) {
  v16h b;
#pragma unroll
  for (int i = 0; i < 16; ++i) b[i] = p[i];
  return b;
}

__device__ __forceinline__ v8f wmma16(v16h a, v16h b, v8f c) {
  return __builtin_amdgcn_wmma_f32_16x16x32_f16(
      /*neg_a=*/false, a, /*neg_b=*/false, b,
      /*c_mod=*/(short)0, c, /*reuse_a=*/false, /*reuse_b=*/false);
}

// ---------------------------------------------------------------------------
// Kernel 0: f32 -> f16 conversion (8 elements per thread).
// ---------------------------------------------------------------------------
__global__ void cvt_f16_kernel(const float* __restrict__ src,
                               _Float16* __restrict__ dst, int n8) {
  const int i = blockIdx.x * blockDim.x + threadIdx.x;
  if (i >= n8) return;
  const float4* s4 = (const float4*)(src + (size_t)i * 8);
  float4 a = s4[0], b = s4[1];
  _Float16 h[8] = {(_Float16)a.x, (_Float16)a.y, (_Float16)a.z, (_Float16)a.w,
                   (_Float16)b.x, (_Float16)b.y, (_Float16)b.z, (_Float16)b.w};
  *(float4*)(dst + (size_t)i * 8) = *(const float4*)h;
}

// ---------------------------------------------------------------------------
// Kernel 1: QKV projection (f16 operands, single-buffered, pointer-stepped).
//   mode 0/1 : out[b*H + h][s][d]   (Q and K)
//   mode 2   : out[b*H + h][d][s]   (V transposed -> contiguous t for PV)
// Grid: (ROWS/16, HIDDEN/256); block = 128 (4 waves, each owns a 16x64 tile).
// Low VGPR pressure (~80) -> many waves in flight hide the load latency.
// ---------------------------------------------------------------------------
__global__ void qkv_proj_kernel(const _Float16* __restrict__ Xh,
                                const _Float16* __restrict__ Wh,
                                const float* __restrict__ bias,
                                _Float16* __restrict__ out, int mode) {
  const int lane = threadIdx.x & 31;
  const int wave = threadIdx.x >> 5;
  const int g   = lane >> 4;
  const int l16 = lane & 15;
  const int r0  = blockIdx.x * 16;
  const int n0  = blockIdx.y * 256 + wave * 64;

  // Per-lane stream pointers, bumped by 32 halves per k-step.
  const _Float16* pa = Xh + (size_t)(r0 + l16) * HIDDEN + 16 * g;
  const _Float16* pb0 = Wh + (size_t)(n0 +  0 + l16) * HIDDEN + 16 * g;
  const _Float16* pb1 = Wh + (size_t)(n0 + 16 + l16) * HIDDEN + 16 * g;
  const _Float16* pb2 = Wh + (size_t)(n0 + 32 + l16) * HIDDEN + 16 * g;
  const _Float16* pb3 = Wh + (size_t)(n0 + 48 + l16) * HIDDEN + 16 * g;

  v8f acc[4] = {};
  for (int kk = 0; kk < HIDDEN; kk += 32) {
    v16h a  = fragA(pa);
    v16h b0 = fragB(pb0);
    v16h b1 = fragB(pb1);
    v16h b2 = fragB(pb2);
    v16h b3 = fragB(pb3);
    acc[0] = wmma16(a, b0, acc[0]);
    acc[1] = wmma16(a, b1, acc[1]);
    acc[2] = wmma16(a, b2, acc[2]);
    acc[3] = wmma16(a, b3, acc[3]);
    pa += 32; pb0 += 32; pb1 += 32; pb2 += 32; pb3 += 32;
  }

#pragma unroll
  for (int nt = 0; nt < 4; ++nt) {
    const int n  = n0 + nt * 16 + l16;
    const float bv = bias[n];
    const int h = n >> 6, d = n & 63;
#pragma unroll
    for (int i = 0; i < 8; ++i) {
      const int R = r0 + i + 8 * g;          // flattened (s,b) row
      const int s = R >> 2, b = R & 3;       // BATCH = 4
      const int bh = b * HEADS + h;
      const float val = acc[nt][i] + bv;
      if (mode == 2)
        out[((size_t)(bh * HEAD_DIM + d) << 10) + s] = (_Float16)val;   // [bh][d][s]
      else
        out[(((size_t)bh << 10) + s) * HEAD_DIM + d] = (_Float16)val;   // [bh][s][d]
    }
  }
}

// ---------------------------------------------------------------------------
// Kernel 2: per-head attention, one wave per (head, 16-row s-block).
// Two-pass with the 16x1024 score strip in LDS as f16 (32 KB -> ~10 resident
// single-wave blocks per 320 KB WGP). Softmax math in f32.
// Grid: (SEQ/16, BATCH*HEADS); block = 32; dynamic LDS = 32 KB + 64 B.
// ---------------------------------------------------------------------------
__global__ void attn_kernel(const _Float16* __restrict__ Q,
                            const _Float16* __restrict__ K,
                            const _Float16* __restrict__ Vt,
                            const float* __restrict__ mask,
                            float* __restrict__ out) {
  extern __shared__ _Float16 smem_h[];
  _Float16* sc = smem_h;                       // [16][1024] scores -> probs (f16)
  float* rinv  = (float*)(smem_h + 16 * SEQ);  // [16] 1/rowsum

  const int lane = threadIdx.x & 31;
  const int g    = lane >> 4;
  const int l16  = lane & 15;
  const int s0   = blockIdx.x * 16;
  const int bh   = blockIdx.y;
  const int b    = bh >> 4, h = bh & 15;

  const _Float16* qh = Q  + (size_t)bh * SEQ * HEAD_DIM;
  const _Float16* kh = K  + (size_t)bh * SEQ * HEAD_DIM;
  const _Float16* vh = Vt + (size_t)bh * HEAD_DIM * SEQ;
  const float* mrow  = mask + (size_t)b * SEQ * SEQ;   // additive mask [s][t]
  const float scale  = 0.125f;                         // 1/sqrt(64)

  // Loop-invariant Q fragments (16 rows x d=0..63)
  v16h qa0 = fragA(qh + (size_t)(s0 + l16) * HEAD_DIM + 16 * g);
  v16h qa1 = fragA(qh + (size_t)(s0 + l16) * HEAD_DIM + 32 + 16 * g);

  // ---- Pass 1: S = scale * Q K^T + mask -> LDS (f16) ----
  const _Float16* pk = kh + (size_t)l16 * HEAD_DIM + 16 * g;  // element (d,t)=K[t][d]
  for (int j = 0; j < SEQ / 16; ++j) {
    const int t0 = j * 16;
    v16h kb0 = fragB(pk);
    v16h kb1 = fragB(pk + 32);
    v8f s8 = {};
    s8 = wmma16(qa0, kb0, s8);
    s8 = wmma16(qa1, kb1, s8);
#pragma unroll
    for (int i = 0; i < 8; ++i) {
      const int row = i + 8 * g;
      const float val = s8[i] * scale + mrow[(size_t)(s0 + row) * SEQ + t0 + l16];
      sc[row * SEQ + t0 + l16] = (_Float16)val;
    }
    pk += 16 * HEAD_DIM;
  }
  __syncthreads();

  // ---- Softmax: lane l handles row l%16 over column half g (f32 math) ----
  {
    const int row = l16;
    const v8h* r8 = (const v8h*)(sc + row * SEQ + g * 512);
    float m = -1e30f;
    for (int c = 0; c < 64; ++c) {
      v8h x = r8[c];
#pragma unroll
      for (int e = 0; e < 8; ++e) m = fmaxf(m, (float)x[e]);
    }
    m = fmaxf(m, __shfl_xor(m, 16));
    float l = 0.f;
    v8h* w8 = (v8h*)(sc + row * SEQ + g * 512);
    for (int c = 0; c < 64; ++c) {
      v8h x = w8[c];
#pragma unroll
      for (int e = 0; e < 8; ++e) {
        const float p = __expf((float)x[e] - m);
        x[e] = (_Float16)p;
        l += p;
      }
      w8[c] = x;
    }
    l += __shfl_xor(l, 16);
    if (g == 0) rinv[row] = 1.0f / l;
  }
  __syncthreads();

  // ---- Pass 2: ctx = P V  (P fragments read straight out of LDS) ----
  const _Float16* pp  = sc + (size_t)l16 * SEQ + 16 * g;   // A-frag stream (LDS)
  const _Float16* pv0 = vh + (size_t)( 0 + l16) * SEQ + 16 * g;  // (t,d)=Vt[d][t]
  const _Float16* pv1 = vh + (size_t)(16 + l16) * SEQ + 16 * g;
  const _Float16* pv2 = vh + (size_t)(32 + l16) * SEQ + 16 * g;
  const _Float16* pv3 = vh + (size_t)(48 + l16) * SEQ + 16 * g;

  v8f o[4] = {};
  for (int t0 = 0; t0 < SEQ; t0 += 32) {
    v16h pa = fragA(pp);
    v16h v0 = fragB(pv0);
    v16h v1 = fragB(pv1);
    v16h v2 = fragB(pv2);
    v16h v3 = fragB(pv3);
    o[0] = wmma16(pa, v0, o[0]);
    o[1] = wmma16(pa, v1, o[1]);
    o[2] = wmma16(pa, v2, o[2]);
    o[3] = wmma16(pa, v3, o[3]);
    pp += 32; pv0 += 32; pv1 += 32; pv2 += 32; pv3 += 32;
  }

  // ---- Epilogue: scale by 1/rowsum, write (S, B, HIDDEN) fp32 ----
#pragma unroll
  for (int dt = 0; dt < 4; ++dt) {
#pragma unroll
    for (int i = 0; i < 8; ++i) {
      const int row = i + 8 * g;
      const int s = s0 + row;
      const int d = dt * 16 + l16;
      out[(size_t)s * (BATCH * HIDDEN) + b * HIDDEN + h * HEAD_DIM + d] =
          o[dt][i] * rinv[row];
    }
  }
}

// ---------------------------------------------------------------------------
extern "C" void kernel_launch(void* const* d_in, const int* in_sizes, int n_in,
                              void* d_out, int out_size, void* d_ws, size_t ws_size,
                              hipStream_t stream) {
  (void)in_sizes; (void)n_in; (void)out_size; (void)ws_size;
  const float* X    = (const float*)d_in[0];
  const float* mask = (const float*)d_in[1];
  const float* Wq   = (const float*)d_in[2];
  const float* bq   = (const float*)d_in[3];
  const float* Wk   = (const float*)d_in[4];
  const float* bk   = (const float*)d_in[5];
  const float* Wv   = (const float*)d_in[6];
  const float* bv   = (const float*)d_in[7];

  // Workspace layout (f16): Q | K | Vt | Xh | WqH | WkH | WvH  = 38 MB total
  _Float16* q   = (_Float16*)d_ws;
  _Float16* k   = q   + (size_t)ROWS * HIDDEN;
  _Float16* vt  = k   + (size_t)ROWS * HIDDEN;
  _Float16* xh  = vt  + (size_t)ROWS * HIDDEN;
  _Float16* wqh = xh  + (size_t)ROWS * HIDDEN;
  _Float16* wkh = wqh + (size_t)HIDDEN * HIDDEN;
  _Float16* wvh = wkh + (size_t)HIDDEN * HIDDEN;

  // Pre-convert inputs to f16 (bandwidth-bound, removes cvt from GEMM loops)
  {
    const int nX = ROWS * HIDDEN / 8, nW = HIDDEN * HIDDEN / 8;
    cvt_f16_kernel<<<(nX + 255) / 256, 256, 0, stream>>>(X, xh, nX);
    cvt_f16_kernel<<<(nW + 255) / 256, 256, 0, stream>>>(Wq, wqh, nW);
    cvt_f16_kernel<<<(nW + 255) / 256, 256, 0, stream>>>(Wk, wkh, nW);
    cvt_f16_kernel<<<(nW + 255) / 256, 256, 0, stream>>>(Wv, wvh, nW);
  }

  dim3 gA(ROWS / 16, HIDDEN / 256, 1);
  qkv_proj_kernel<<<gA, 128, 0, stream>>>(xh, wqh, bq, q, 0);
  qkv_proj_kernel<<<gA, 128, 0, stream>>>(xh, wkh, bk, k, 1);
  qkv_proj_kernel<<<gA, 128, 0, stream>>>(xh, wvh, bv, vt, 2);

  dim3 gB(SEQ / 16, BATCH * HEADS, 1);
  const size_t lds_bytes = (size_t)16 * SEQ * sizeof(_Float16) + 16 * sizeof(float);
  attn_kernel<<<gB, 32, lds_bytes, stream>>>(q, k, vt, mask, (float*)d_out);
}
